// MultiHeadSelfAttn_87892210746040
// MI455X (gfx1250) — matrix operand order repaired and checked by reading, not compiled, not run on hardware
//
#include <hip/hip_runtime.h>
#include <hip/hip_bf16.h>

// ---------------------------------------------------------------------------
// Multi-head self-attention forward for MI455X (gfx1250), wave32 + WMMA.
//   outputs: attn [B,NH,S,S] f32  (512 MB -> dominates; written exactly once)
//            out  [B,S,OUT]  f32
// f16 WMMA (f32 accumulate) everywhere. Score tile (32 x 2048 f32) is
// LDS-resident in the 320KB WGP LDS, so softmax never touches HBM.
// B operands staged TRANSPOSED ([col][k]) -> fragments load as ds_load_b128.
// K/Q staging uses GLOBAL_LOAD_ASYNC_TO_LDS_B128 (ASYNCcnt path).
// ---------------------------------------------------------------------------

#define B_    2
#define S_    2048
#define INP_  1024
#define HID_  64
#define NH_   16
#define OUT_  1024
#define NEG_  (-1e26f)

typedef _Float16 half_t;
typedef __attribute__((ext_vector_type(16))) _Float16 v16h;
typedef __attribute__((ext_vector_type(8)))  float    v8f;

// ---- WMMA fragment helpers (layouts per CDNA5 ISA 7.12.2) ------------------

__device__ __forceinline__ v8f wmma_f16(v16h a, v16h b, v8f c) {
  return __builtin_amdgcn_wmma_f32_16x16x32_f16(
      /*neg_a=*/false, a, /*neg_b=*/false, b,
      /*c_mod=*/(short)0, c, /*reuse_a=*/false, /*reuse_b=*/false);
}

// A-matrix 16x32 f16 from row-major [m][k]: lane L holds row M=L&15, g=L>>4;
// element runs are contiguous (K = 2p+8g pattern) -> merges to ds_load_b128.
__device__ __forceinline__ v16h load_a16(const half_t* s, int ld, int row0, int k0) {
  const int lane = threadIdx.x & 31;
  const int g = lane >> 4;
  const half_t* p = s + (size_t)(row0 + (lane & 15)) * ld;
  v16h a;
#pragma unroll
  for (int i = 0; i < 8; ++i) {
    const int kb = k0 + ((i < 4) ? (2 * i + 8 * g) : (16 + 2 * (i - 4) + 8 * g));
    a[2 * i]     = p[kb];
    a[2 * i + 1] = p[kb + 1];
  }
  return a;
}

// Same A layout, f32 LDS source (softmax probabilities), converted to f16.
__device__ __forceinline__ v16h load_a16_f32(const float* s, int ld, int row0, int k0) {
  const int lane = threadIdx.x & 31;
  const int g = lane >> 4;
  const float* p = s + (size_t)(row0 + (lane & 15)) * ld;
  v16h a;
#pragma unroll
  for (int i = 0; i < 8; ++i) {
    const int kb = k0 + ((i < 4) ? (2 * i + 8 * g) : (16 + 2 * (i - 4) + 8 * g));
    a[2 * i]     = (half_t)p[kb];
    a[2 * i + 1] = (half_t)p[kb + 1];
  }
  return a;
}

// B-matrix 32x16 f16 from a TRANSPOSED stage [col][k]: lane col N = lane&15,
// K = 8g + 0..7 and 16 + 8g + 0..7 -> two ds_load_b128 per fragment.
__device__ __forceinline__ v16h load_b16T(const half_t* s, int ld, int k0, int col0) {
  const int lane = threadIdx.x & 31;
  const int g = lane >> 4;
  const half_t* p = s + (size_t)(col0 + (lane & 15)) * ld + k0 + 8 * g;
  v16h b;
#pragma unroll
  for (int i = 0; i < 16; ++i) b[i] = p[(i < 8) ? i : (i + 8)];
  return b;
}

// ---- CDNA5 async global->LDS copy (ASYNCcnt path, ISA 15.18.3 op 98) -------
// VDST = VGPR with LDS byte address (= addr[31:0] of the generic pointer per
// the LDS aperture mapping), VADDR = 64-bit global address, SADDR = off.
__device__ __forceinline__ void async_copy_b128(void* lds_dst, const void* gsrc) {
  const unsigned lds = (unsigned)(size_t)lds_dst;
  asm volatile("global_load_async_to_lds_b128 %0, %1, off"
               :: "v"(lds), "v"((unsigned long long)(size_t)gsrc)
               : "memory");
}
__device__ __forceinline__ void wait_async0() {
  asm volatile("s_wait_asynccnt 0" ::: "memory");
}

// ---------------------------------------------------------------------------
// Kernel 1: Q/K/V projection.  C = x @ W + b, stored f16 head-split
// [b][head][s][h], head = col & 15, h = col >> 4   (reshape(b,s,HID,NH)).
// Tile: 128(M) x 64(N), BK=64, 8 waves of 32x32.
// ---------------------------------------------------------------------------
__global__ __launch_bounds__(256) void qkv_proj_kernel(
    const float* __restrict__ x, const float* __restrict__ W,
    const float* __restrict__ bias, half_t* __restrict__ dst) {
  __shared__ half_t sA[128][72];    // [m][k]
  __shared__ half_t sBt[64][72];    // [n][k]  (transposed W tile)

  const int m0 = blockIdx.y * 128;
  const int c0 = blockIdx.x * 64;
  const int tid = threadIdx.x;
  const int lane = tid & 31;
  const int w = tid >> 5;
  const int wm = (w & 3) * 32;   // wave M offset
  const int wn = (w >> 2) * 32;  // wave N offset

  v8f acc[2][2];
#pragma unroll
  for (int i = 0; i < 2; ++i)
#pragma unroll
    for (int j = 0; j < 2; ++j) acc[i][j] = v8f{};

  for (int k0 = 0; k0 < INP_; k0 += 64) {
    // Stage x tile (128x64 f32 -> f16), [m][k]
#pragma unroll
    for (int i = 0; i < 8; ++i) {
      const int idx = tid + i * 256;         // 0..2047
      const int r = idx >> 4;
      const int cc = (idx & 15) * 4;
      const float4 v = *(const float4*)(x + (size_t)(m0 + r) * INP_ + k0 + cc);
      sA[r][cc + 0] = (half_t)v.x;
      sA[r][cc + 1] = (half_t)v.y;
      sA[r][cc + 2] = (half_t)v.z;
      sA[r][cc + 3] = (half_t)v.w;
    }
    // Stage W tile (64x64 f32 -> f16), TRANSPOSED to [n][k]
#pragma unroll
    for (int i = 0; i < 4; ++i) {
      const int idx = tid + i * 256;         // 0..1023
      const int r = idx >> 4;                // k row
      const int cc = (idx & 15) * 4;         // n col
      const float4 v = *(const float4*)(W + (size_t)(k0 + r) * OUT_ + c0 + cc);
      sBt[cc + 0][r] = (half_t)v.x;
      sBt[cc + 1][r] = (half_t)v.y;
      sBt[cc + 2][r] = (half_t)v.z;
      sBt[cc + 3][r] = (half_t)v.w;
    }
    __syncthreads();
#pragma unroll
    for (int kk = 0; kk < 64; kk += 32) {
      const v16h a0 = load_a16(&sA[0][0], 72, wm + 0,  kk);
      const v16h a1 = load_a16(&sA[0][0], 72, wm + 16, kk);
      const v16h b0 = load_b16T(&sBt[0][0], 72, kk, wn + 0);
      const v16h b1 = load_b16T(&sBt[0][0], 72, kk, wn + 16);
      acc[0][0] = wmma_f16(a0, b0, acc[0][0]);
      acc[0][1] = wmma_f16(a0, b1, acc[0][1]);
      acc[1][0] = wmma_f16(a1, b0, acc[1][0]);
      acc[1][1] = wmma_f16(a1, b1, acc[1][1]);
    }
    __syncthreads();
  }

  // Scatter to [b][head][s][h] f16 with bias.
#pragma unroll
  for (int msub = 0; msub < 2; ++msub) {
#pragma unroll
    for (int nsub = 0; nsub < 2; ++nsub) {
      const int col  = c0 + wn + nsub * 16 + (lane & 15);
      const int head = col & (NH_ - 1);
      const int h    = col >> 4;
      const float bi = bias[col];
#pragma unroll
      for (int vr = 0; vr < 8; ++vr) {
        const int m = m0 + wm + msub * 16 + vr + 8 * (lane >> 4);
        const int b = m >> 11;            // m / S_
        const int s = m & (S_ - 1);
        dst[(((size_t)b * NH_ + head) * S_ + s) * HID_ + h] =
            (half_t)(acc[msub][nsub][vr] + bi);
      }
    }
  }
}

// ---------------------------------------------------------------------------
// Kernel 2: attention for one (b, head, 32-query tile).
// ---------------------------------------------------------------------------
#define SLD (S_ + 8)   // padded score stride (floats); SLD*4 bytes % 16 == 0

__global__ __launch_bounds__(256) void attn_kernel(
    const half_t* __restrict__ Qh, const half_t* __restrict__ Kh,
    const half_t* __restrict__ Vh,
    const unsigned char* __restrict__ pad_mask,
    const unsigned char* __restrict__ attn_mask,
    float* __restrict__ attn_out, half_t* __restrict__ ctx) {
  extern __shared__ char smem[];
  float*  sc   = (float*)smem;                              // [32][SLD]
  half_t* qt   = (half_t*)(smem + (size_t)32 * SLD * 4);    // [32][64]
  half_t* kvst = qt + 32 * 64;            // K phase: [64][64]; V phase: [64][72]
  unsigned char* mbuf = (unsigned char*)(kvst + 64 * 72);   // [32][64]
  __shared__ float rmax[32], rsum[32];

  const int q0   = blockIdx.x * 32;
  const int head = blockIdx.y;
  const int b    = blockIdx.z;
  const int tid  = threadIdx.x;
  const int lane = tid & 31;
  const int w    = tid >> 5;
  const size_t base = ((size_t)(b * NH_ + head)) * S_ * HID_;

  // ---- stage Q tile (async global->LDS) and build per-wave A fragments ----
  async_copy_b128(&qt[tid * 8], Qh + base + (size_t)q0 * HID_ + tid * 8);
  wait_async0();
  __syncthreads();

  const int qh = (w & 1) * 16;   // wave's query sub-block (0 or 16)
  const int wk = (w >> 1) * 16;  // wave's key sub-offset / PV h-tile
  const v16h aQ0 = load_a16(qt, 64, qh, 0);
  const v16h aQ1 = load_a16(qt, 64, qh, 32);

  // pad_mask applies per query row only: hoist to 8 per-lane flags.
  unsigned char prow[8];
#pragma unroll
  for (int vr = 0; vr < 8; ++vr)
    prow[vr] = pad_mask[(size_t)b * S_ + q0 + qh + vr + 8 * (lane >> 4)];

  // ---- raw scores + masks into LDS ----
  for (int kc = 0; kc < S_; kc += 64) {
    __syncthreads();  // protect stage buffers from previous chunk's readers
    // K chunk [key][h], contiguous async copy (8 KB)
#pragma unroll
    for (int i = 0; i < 2; ++i) {
      const int idx = tid + i * 256;       // 0..511
      async_copy_b128(&kvst[idx * 8], Kh + base + (size_t)kc * HID_ + idx * 8);
    }
    // attn_mask tile 32 x 64 bytes (uint4)
    if (tid < 128) {
      const int r = tid >> 2;
      const int cc = (tid & 3) * 16;
      *(uint4*)&mbuf[r * 64 + cc] =
          *(const uint4*)(attn_mask + (size_t)(q0 + r) * S_ + kc + cc);
    }
    if (kc + 64 < S_)  // pull next chunk toward cache (global_prefetch_b8)
      __builtin_prefetch((const void*)(Kh + base + (size_t)(kc + 64) * HID_ + tid * 32), 0, 1);
    wait_async0();
    __syncthreads();

    v8f accS{};
    accS = wmma_f16(aQ0, load_b16T(kvst, 64, 0,  wk), accS);
    accS = wmma_f16(aQ1, load_b16T(kvst, 64, 32, wk), accS);

    const int kl = wk + (lane & 15);       // key col within chunk
#pragma unroll
    for (int vr = 0; vr < 8; ++vr) {
      const int qrow = qh + vr + 8 * (lane >> 4);
      float v = accS[vr];
      if (prow[vr])             v = NEG_;
      if (mbuf[qrow * 64 + kl]) v = NEG_;
      sc[(size_t)qrow * SLD + kc + kl] = v;
    }
  }
  __syncthreads();

  // ---- softmax: each row handled by an 8-lane group, float4 sweeps ----
  {
    const int r = tid >> 3, c8 = tid & 7;
    float mx = -3.4e38f;
    for (int c = c8 * 4; c < S_; c += 32) {
      const float4 v = *(const float4*)&sc[(size_t)r * SLD + c];
      mx = fmaxf(mx, fmaxf(fmaxf(v.x, v.y), fmaxf(v.z, v.w)));
    }
    mx = fmaxf(mx, __shfl_xor(mx, 1));
    mx = fmaxf(mx, __shfl_xor(mx, 2));
    mx = fmaxf(mx, __shfl_xor(mx, 4));
    if (c8 == 0) rmax[r] = mx;
  }
  __syncthreads();
  {
    const int r = tid >> 3, c8 = tid & 7;
    const float mx = rmax[r];
    float sm = 0.f;
    for (int c = c8 * 4; c < S_; c += 32) {
      float4 v = *(const float4*)&sc[(size_t)r * SLD + c];
      v.x = __expf((v.x - mx) * 0.125f);   // 1/sqrt(HID) folded into exp
      v.y = __expf((v.y - mx) * 0.125f);
      v.z = __expf((v.z - mx) * 0.125f);
      v.w = __expf((v.w - mx) * 0.125f);
      *(float4*)&sc[(size_t)r * SLD + c] = v;
      sm += v.x + v.y + v.z + v.w;
    }
    sm += __shfl_xor(sm, 1);
    sm += __shfl_xor(sm, 2);
    sm += __shfl_xor(sm, 4);
    if (c8 == 0) rsum[r] = sm;
  }
  __syncthreads();

  // ---- normalize in LDS and write attn (coalesced float4, once) ----
  float* aout = attn_out + ((size_t)(b * NH_ + head) * S_ + q0) * S_;
  for (int r = 0; r < 32; ++r) {
    const float inv = 1.f / rsum[r];
#pragma unroll
    for (int i = 0; i < 2; ++i) {
      const int c = (tid + i * 256) * 4;
      float4 v = *(const float4*)&sc[(size_t)r * SLD + c];
      v.x *= inv; v.y *= inv; v.z *= inv; v.w *= inv;
      *(float4*)&sc[(size_t)r * SLD + c] = v;
      *(float4*)&aout[(size_t)r * S_ + c] = v;
    }
  }
  __syncthreads();

  // ---- P @ V ----
  v8f accO{};
  half_t* vt = kvst;                       // reuse stage region as [h][key+pad]
  for (int kc = 0; kc < S_; kc += 64) {
    __syncthreads();
    // V chunk transposed: global [key][h] -> LDS vt[h][key] (ld 72)
#pragma unroll
    for (int i = 0; i < 2; ++i) {
      const int idx = tid + i * 256;       // 0..511
      const int key = idx >> 3;
      const int h0 = (idx & 7) * 8;
      const uint4 v = *(const uint4*)(Vh + base + (size_t)(kc + key) * HID_ + h0);
      const half_t* hv = (const half_t*)&v;
#pragma unroll
      for (int j = 0; j < 8; ++j) vt[(h0 + j) * 72 + key] = hv[j];
    }
    if (kc + 64 < S_)
      __builtin_prefetch((const void*)(Vh + base + (size_t)(kc + 64) * HID_ + tid * 32), 0, 1);
    __syncthreads();
#pragma unroll
    for (int sub = 0; sub < 2; ++sub) {
      const v16h aP = load_a16_f32(sc, SLD, qh, kc + sub * 32);
      const v16h bV = load_b16T(vt, 72, sub * 32, wk);   // B rows = key, cols = h
      accO = wmma_f16(aP, bV, accO);
    }
  }

  // ctx layout [b][s][head*HID + h] f16 (reference's heads-major flatten)
  const int h = wk + (lane & 15);
#pragma unroll
  for (int vr = 0; vr < 8; ++vr) {
    const int qg = q0 + qh + vr + 8 * (lane >> 4);
    ctx[((size_t)b * S_ + qg) * (NH_ * HID_) + head * HID_ + h] = (half_t)accO[vr];
  }
}

// ---------------------------------------------------------------------------
// Kernel 3: out = ctx @ W_O + b_O  (f32 output, row-major).
// ---------------------------------------------------------------------------
__global__ __launch_bounds__(256) void out_proj_kernel(
    const half_t* __restrict__ ctx, const float* __restrict__ W,
    const float* __restrict__ bias, float* __restrict__ out) {
  __shared__ half_t sA[128][80];   // [m][k]; stride 160B keeps uint4 stores aligned
  __shared__ half_t sBt[64][72];   // [n][k]

  const int m0 = blockIdx.y * 128;
  const int c0 = blockIdx.x * 64;
  const int tid = threadIdx.x;
  const int lane = tid & 31;
  const int w = tid >> 5;
  const int wm = (w & 3) * 32;
  const int wn = (w >> 2) * 32;

  v8f acc[2][2];
#pragma unroll
  for (int i = 0; i < 2; ++i)
#pragma unroll
    for (int j = 0; j < 2; ++j) acc[i][j] = v8f{};

  const int K = NH_ * HID_;  // 1024
  for (int k0 = 0; k0 < K; k0 += 64) {
    // A tile: f16 direct async copy to LDS
#pragma unroll
    for (int i = 0; i < 4; ++i) {
      const int idx = tid + i * 256;       // 0..1023
      const int r = idx >> 3;
      const int cc = (idx & 7) * 8;
      async_copy_b128(&sA[r][cc], ctx + (size_t)(m0 + r) * K + k0 + cc);
    }
    // B tile: f32 -> f16, TRANSPOSED to [n][k]
#pragma unroll
    for (int i = 0; i < 4; ++i) {
      const int idx = tid + i * 256;
      const int r = idx >> 4;                // k row
      const int cc = (idx & 15) * 4;         // n col
      const float4 v = *(const float4*)(W + (size_t)(k0 + r) * OUT_ + c0 + cc);
      sBt[cc + 0][r] = (half_t)v.x;
      sBt[cc + 1][r] = (half_t)v.y;
      sBt[cc + 2][r] = (half_t)v.z;
      sBt[cc + 3][r] = (half_t)v.w;
    }
    wait_async0();
    __syncthreads();
#pragma unroll
    for (int kk = 0; kk < 64; kk += 32) {
      const v16h a0 = load_a16(&sA[0][0], 80, wm + 0,  kk);
      const v16h a1 = load_a16(&sA[0][0], 80, wm + 16, kk);
      const v16h b0 = load_b16T(&sBt[0][0], 72, kk, wn + 0);
      const v16h b1 = load_b16T(&sBt[0][0], 72, kk, wn + 16);
      acc[0][0] = wmma_f16(a0, b0, acc[0][0]);
      acc[0][1] = wmma_f16(a0, b1, acc[0][1]);
      acc[1][0] = wmma_f16(a1, b0, acc[1][0]);
      acc[1][1] = wmma_f16(a1, b1, acc[1][1]);
    }
    __syncthreads();
  }

#pragma unroll
  for (int msub = 0; msub < 2; ++msub) {
#pragma unroll
    for (int nsub = 0; nsub < 2; ++nsub) {
      const int col = c0 + wn + nsub * 16 + (lane & 15);
      const float bi = bias[col];
#pragma unroll
      for (int vr = 0; vr < 8; ++vr) {
        const int m = m0 + wm + msub * 16 + vr + 8 * (lane >> 4);
        out[(size_t)m * OUT_ + col] = acc[msub][nsub][vr] + bi;
      }
    }
  }
}

// ---------------------------------------------------------------------------
// Launch. Inputs (setup_inputs order): x, pad_mask, attn_mask,
// W_Q,b_Q, W_K,b_K, W_V,b_V, W_O,b_O.  d_out = [attn | out] f32.
// Workspace: Qh,Kh,Vh,ctx f16 = 32 MB.
// ---------------------------------------------------------------------------
extern "C" void kernel_launch(void* const* d_in, const int* in_sizes, int n_in,
                              void* d_out, int out_size, void* d_ws, size_t ws_size,
                              hipStream_t stream) {
  const float*         x   = (const float*)d_in[0];
  const unsigned char* pad = (const unsigned char*)d_in[1];   // bool
  const unsigned char* am  = (const unsigned char*)d_in[2];   // bool [1,1,S,S]
  const float* W_Q = (const float*)d_in[3];
  const float* b_Q = (const float*)d_in[4];
  const float* W_K = (const float*)d_in[5];
  const float* b_K = (const float*)d_in[6];
  const float* W_V = (const float*)d_in[7];
  const float* b_V = (const float*)d_in[8];
  const float* W_O = (const float*)d_in[9];
  const float* b_O = (const float*)d_in[10];

  const size_t QKV = (size_t)B_ * NH_ * S_ * HID_;   // 4,194,304 halves each
  half_t* Qh  = (half_t*)d_ws;
  half_t* Kh  = Qh + QKV;
  half_t* Vh  = Kh + QKV;
  half_t* ctx = Vh + QKV;

  float* attn = (float*)d_out;
  float* out  = attn + (size_t)B_ * NH_ * S_ * S_;

  const dim3 gProj(OUT_ / 64, (B_ * S_) / 128, 1);   // (16, 32)
  qkv_proj_kernel<<<gProj, 256, 0, stream>>>(x, W_Q, b_Q, Qh);
  qkv_proj_kernel<<<gProj, 256, 0, stream>>>(x, W_K, b_K, Kh);
  qkv_proj_kernel<<<gProj, 256, 0, stream>>>(x, W_V, b_V, Vh);

  const size_t smem = (size_t)32 * SLD * 4   // score tile
                    + (size_t)32 * 64 * 2    // Q tile
                    + (size_t)64 * 72 * 2    // K/V stage (overlaid)
                    + (size_t)32 * 64;       // attn_mask tile   (~280 KB < 320 KB)
  attn_kernel<<<dim3(S_ / 32, NH_, B_), 256, smem, stream>>>(
      Qh, Kh, Vh, pad, am, attn, ctx);

  out_proj_kernel<<<gProj, 256, 0, stream>>>(ctx, W_O, b_O, out);
}
